// IB_FactorizedPrior_25262997635225
// MI455X (gfx1250) — matrix operand order repaired and checked by reading, not compiled, not run on hardware
//
#include <hip/hip_runtime.h>

// ---------------------------------------------------------------------------
// Problem constants (reference: B=4096, C=1000, D=3072)
// ---------------------------------------------------------------------------
#define BN    4096          // batch rows
#define CN    1000          // valid classes
#define CP    1024          // padded classes (64 tiles of 16)
#define DN    3072          // feature dim
#define NKT   (DN / 32)     // 96 K-tiles of 32
#define NRT   (BN / 16)     // 256 row tiles
#define NCT   (CP / 16)     // 64 class tiles
#define NCHUNK 4            // class chunks (16 class-tiles each)

typedef __attribute__((ext_vector_type(16))) __bf16 v16bf;
typedef __attribute__((ext_vector_type(8)))  float  v8f;

// ---------------------------------------------------------------------------
// fp32 -> (hi, lo) bf16 split with round-to-nearest-even
// ---------------------------------------------------------------------------
__device__ __forceinline__ unsigned short f32_to_bf16_rne(float x) {
    unsigned u = __float_as_uint(x);
    unsigned r = u + 0x7FFFu + ((u >> 16) & 1u);
    return (unsigned short)(r >> 16);
}
__device__ __forceinline__ float bf16_bits_to_f32(unsigned short h) {
    return __uint_as_float(((unsigned)h) << 16);
}

// ---------------------------------------------------------------------------
// Split a row-major fp32 matrix into hi/lo bf16 tiles stored in WMMA fragment
// order: tile (r,t) -> 32 lanes x 16 halfs contiguous.
//   A layout (16-bit A 16x32, ISA 7.12.2): VGPR v<4: K=2v,2v+1 ; v>=4: K=16+2(v-4)
//     (+8 for lanes 16..31)
//   B layout (32x16): same K order as A: v<4: K=2v,2v+1 ; v>=4: K=16+2(v-4)
//     (+8 for lanes 16..31)
// One wave per tile; 4 waves per block.
// ---------------------------------------------------------------------------
__global__ __launch_bounds__(128) void split_swizzle_kernel(
    const float* __restrict__ src, unsigned short* __restrict__ dhi,
    unsigned short* __restrict__ dlo, int validRows, int cols, int nkt,
    int ntiles, int bLayout)
{
    int wid = blockIdx.x * (blockDim.x >> 5) + (threadIdx.x >> 5);
    if (wid >= ntiles) return;
    int lane = threadIdx.x & 31;
    int r = wid / nkt;
    int t = wid % nkt;
    int row = r * 16 + (lane & 15);
    int laneHi = (lane >> 4);   // 0 or 1

    unsigned short hh[16] __attribute__((aligned(16)));
    unsigned short hl[16] __attribute__((aligned(16)));

    #pragma unroll
    for (int v = 0; v < 8; ++v) {
        int kin;
        if (bLayout) kin = ((v < 4) ? 2 * v : 16 + 2 * (v - 4)) + laneHi * 8; // B frag
        else         kin = ((v < 4) ? 2 * v : 16 + 2 * (v - 4)) + laneHi * 8; // A frag
        #pragma unroll
        for (int off = 0; off < 2; ++off) {
            int K = t * 32 + kin + off;
            float x = (row < validRows) ? src[(size_t)row * cols + K] : 0.0f;
            unsigned short h = f32_to_bf16_rne(x);
            float lo = x - bf16_bits_to_f32(h);
            hh[2 * v + off] = h;
            hl[2 * v + off] = f32_to_bf16_rne(lo);
        }
    }
    size_t base = ((size_t)wid * 32 + lane) * 16;   // in ushorts (32B aligned)
    *(uint4*)(dhi + base)     = *(const uint4*)(hh);
    *(uint4*)(dhi + base + 8) = *(const uint4*)(hh + 8);
    *(uint4*)(dlo + base)     = *(const uint4*)(hl);
    *(uint4*)(dlo + base + 8) = *(const uint4*)(hl + 8);
}

// ---------------------------------------------------------------------------
// Per-row sum of squares. One block per row. Rows >= validRows get padVal.
// ---------------------------------------------------------------------------
__global__ __launch_bounds__(256) void sumsq_kernel(
    const float* __restrict__ src, float* __restrict__ out,
    int cols, int validRows, float padVal)
{
    __shared__ float sm[256];
    int row = blockIdx.x;
    float s = 0.0f;
    if (row < validRows) {
        for (int k = threadIdx.x; k < cols; k += blockDim.x) {
            float x = src[(size_t)row * cols + k];
            s += x * x;
        }
    }
    sm[threadIdx.x] = s;
    __syncthreads();
    for (int o = 128; o > 0; o >>= 1) {
        if ((int)threadIdx.x < o) sm[threadIdx.x] += sm[threadIdx.x + o];
        __syncthreads();
    }
    if (threadIdx.x == 0) out[row] = (row < validRows) ? sm[0] : padVal;
}

// ---------------------------------------------------------------------------
// Fused split-bf16 WMMA GEMM + online logsumexp + label gather.
// One wave = one 16-row tile x one 256-class chunk. 1024 waves total.
//
// Key structure: each lane keeps a PRIVATE online LSE over the 16 classes it
// owns (one per class tile) -- pure VALU, no cross-lane traffic in the hot
// loop. A single 16-lane LSE-merge butterfly runs once at the end.
// ---------------------------------------------------------------------------
__global__ __launch_bounds__(256) void gemm_lse_kernel(
    const unsigned short* __restrict__ zhi, const unsigned short* __restrict__ zlo,
    const unsigned short* __restrict__ mhi, const unsigned short* __restrict__ mlo,
    const float* __restrict__ z2, const float* __restrict__ m2,
    const int* __restrict__ labels,
    float* __restrict__ pmax, float* __restrict__ psum,
    float* __restrict__ picked)
{
    const int lane  = threadIdx.x & 31;
    const int gw    = blockIdx.x * (blockDim.x >> 5) + (threadIdx.x >> 5);
    const int rTile = gw >> 2;            // 0..255
    const int chunk = gw & 3;             // 0..3
    const int rBase = rTile << 4;
    const int hi8   = (lane >> 4) << 3;   // rows 0..7 vs 8..15
    const int nCol  = lane & 15;

    float zz[8];
    int   lab[8];
    #pragma unroll
    for (int i = 0; i < 8; ++i) {
        int row = rBase + hi8 + i;
        zz[i]  = z2[row];
        lab[i] = labels[row];
    }

    float rmax[8], rsum[8], pk[8];
    bool  hit[8];
    #pragma unroll
    for (int i = 0; i < 8; ++i) {
        rmax[i] = -INFINITY; rsum[i] = 0.0f; pk[i] = 0.0f; hit[i] = false;
    }

    for (int ctl = 0; ctl < NCT / NCHUNK; ++ctl) {
        const int ct  = chunk * (NCT / NCHUNK) + ctl;   // class tile 0..63
        const int cls = ct * 16 + nCol;                 // this lane's class
        const float mm = m2[cls];

        v8f acc = {};
        #pragma unroll 2
        for (int t = 0; t < NKT; ++t) {
            size_t aoff = ((size_t)(rTile * NKT + t) * 32 + lane) * 16;
            size_t boff = ((size_t)(ct    * NKT + t) * 32 + lane) * 16;
            v16bf ah = *(const v16bf*)(const void*)(zhi + aoff);
            v16bf al = *(const v16bf*)(const void*)(zlo + aoff);
            v16bf bh = *(const v16bf*)(const void*)(mhi + boff);
            v16bf bl = *(const v16bf*)(const void*)(mlo + boff);
            acc = __builtin_amdgcn_wmma_f32_16x16x32_bf16(
                false, ah, false, bh, (short)0, acc, false, false);
            acc = __builtin_amdgcn_wmma_f32_16x16x32_bf16(
                false, ah, false, bl, (short)0, acc, false, false);
            acc = __builtin_amdgcn_wmma_f32_16x16x32_bf16(
                false, al, false, bh, (short)0, acc, false, false);
        }

        // Per-lane epilogue: dists = cross - 0.5*(|z|^2 + |mu|^2);
        // private online LSE update, no cross-lane ops.
        #pragma unroll
        for (int i = 0; i < 8; ++i) {
            float d = acc[i] - 0.5f * (zz[i] + mm);
            bool  h = (cls == lab[i]);
            pk[i]  = h ? d : pk[i];
            hit[i] = hit[i] || h;
            float nm = fmaxf(rmax[i], d);
            rsum[i]  = rsum[i] * __expf(rmax[i] - nm) + __expf(d - nm);
            rmax[i]  = nm;
        }
    }

    // One-time 16-lane LSE merge (xor masks < 16 stay within each half-wave).
    #pragma unroll
    for (int i = 0; i < 8; ++i) {
        float m = rmax[i], s = rsum[i];
        #pragma unroll
        for (int mk = 1; mk <= 8; mk <<= 1) {
            float om = __shfl_xor(m, mk, 32);
            float os = __shfl_xor(s, mk, 32);
            float nm = fmaxf(m, om);
            s = s * __expf(m - nm) + os * __expf(om - nm);
            m = nm;
        }
        rmax[i] = m; rsum[i] = s;
    }

    if (nCol == 0) {
        #pragma unroll
        for (int i = 0; i < 8; ++i) {
            int row = rBase + hi8 + i;
            pmax[row * NCHUNK + chunk] = rmax[i];
            psum[row * NCHUNK + chunk] = rsum[i];
        }
    }
    #pragma unroll
    for (int i = 0; i < 8; ++i) {
        if (hit[i]) picked[rBase + hi8 + i] = pk[i];   // exactly one lane/chunk
    }
}

// ---------------------------------------------------------------------------
// Combine chunk partials -> log_pz; reduce losses over the batch.
// out = { total, loss_gen, loss_cls }
// ---------------------------------------------------------------------------
__global__ __launch_bounds__(256) void finalize_kernel(
    const float* __restrict__ pmax, const float* __restrict__ psum,
    const float* __restrict__ picked, const float* __restrict__ sldj,
    const float* __restrict__ beta, float* __restrict__ out)
{
    __shared__ float s1[256], s2[256];
    float a1 = 0.0f, a2 = 0.0f;
    for (int row = threadIdx.x; row < BN; row += 256) {
        float M = pmax[row * NCHUNK];
        #pragma unroll
        for (int c = 1; c < NCHUNK; ++c) M = fmaxf(M, pmax[row * NCHUNK + c]);
        float S = 0.0f;
        #pragma unroll
        for (int c = 0; c < NCHUNK; ++c)
            S += psum[row * NCHUNK + c] * __expf(pmax[row * NCHUNK + c] - M);
        float lpz = M + __logf(S);
        a1 += lpz - picked[row];
        a2 += lpz + sldj[row];
    }
    s1[threadIdx.x] = a1;
    s2[threadIdx.x] = a2;
    __syncthreads();
    for (int o = 128; o > 0; o >>= 1) {
        if ((int)threadIdx.x < o) {
            s1[threadIdx.x] += s1[threadIdx.x + o];
            s2[threadIdx.x] += s2[threadIdx.x + o];
        }
        __syncthreads();
    }
    if (threadIdx.x == 0) {
        float loss_cls = s1[0] / (float)BN;
        float loss_gen = -s2[0] / ((float)BN * (float)DN);
        out[0] = loss_gen + beta[0] * loss_cls;
        out[1] = loss_gen;
        out[2] = loss_cls;
    }
}

// ---------------------------------------------------------------------------
// Host-side launch
// ---------------------------------------------------------------------------
extern "C" void kernel_launch(void* const* d_in, const int* in_sizes, int n_in,
                              void* d_out, int out_size, void* d_ws, size_t ws_size,
                              hipStream_t stream) {
    const float* z      = (const float*)d_in[0];   // [B, D]
    const float* sldj   = (const float*)d_in[1];   // [B]
    const int*   labels = (const int*)d_in[2];     // [B]
    const float* beta   = (const float*)d_in[3];   // scalar
    const float* means  = (const float*)d_in[4];   // [C, D]
    float*       out    = (float*)d_out;           // {total, gen, cls}

    // Workspace layout (bytes)
    char* w = (char*)d_ws;
    unsigned short* zhi = (unsigned short*)(w);                       // 25165824
    unsigned short* zlo = (unsigned short*)(w + 25165824);            // 25165824
    unsigned short* mhi = (unsigned short*)(w + 50331648);            //  6291456
    unsigned short* mlo = (unsigned short*)(w + 56623104);            //  6291456
    float*          z2p = (float*)(w + 62914560);                     //    16384
    float*          m2p = (float*)(w + 62930944);                     //     4096
    float*          pck = (float*)(w + 62935040);                     //    16384
    float*          pmx = (float*)(w + 62951424);                     //    65536
    float*          psm = (float*)(w + 63016960);                     //    65536

    // 1) Split + swizzle into WMMA fragment order
    {
        int ntiles = NRT * NKT;            // 24576 (A tiles for z)
        split_swizzle_kernel<<<ntiles / 4, 128, 0, stream>>>(
            z, zhi, zlo, BN, DN, NKT, ntiles, /*bLayout=*/0);
    }
    {
        int ntiles = NCT * NKT;            // 6144 (B tiles for means, padded)
        split_swizzle_kernel<<<ntiles / 4, 128, 0, stream>>>(
            means, mhi, mlo, CN, DN, NKT, ntiles, /*bLayout=*/1);
    }

    // 2) Row norms
    sumsq_kernel<<<BN, 256, 0, stream>>>(z, z2p, DN, BN, 0.0f);
    sumsq_kernel<<<CP, 256, 0, stream>>>(means, m2p, DN, CN, 1e30f);

    // 3) Fused WMMA GEMM + online logsumexp (1024 waves, 8 waves/block)
    gemm_lse_kernel<<<(NRT * NCHUNK) / 8, 256, 0, stream>>>(
        zhi, zlo, mhi, mlo, z2p, m2p, labels, pmx, psm, pck);

    // 4) Final loss reduction
    finalize_kernel<<<1, 256, 0, stream>>>(pmx, psm, pck, sldj, beta, out);
}
